// NoriSLMCell_76235669504536
// MI455X (gfx1250) — compile-verified, matrix operand order checked by reading, not run on hardware
//
#include <hip/hip_runtime.h>
#include <math.h>

typedef __bf16 bf16;
typedef __attribute__((ext_vector_type(16))) __bf16 v16bf;
typedef __attribute__((ext_vector_type(8)))  float  v8f;
typedef int b128v __attribute__((vector_size(16)));

#define D_MODEL 512
#define NT 32   // D_MODEL / 16
#define INV_SQRT_D 0.044194173824159216f

// Async global->LDS staging (CDNA5). Guarded so absence of the builtin
// falls back to a plain VGPR round-trip copy.
#if __has_builtin(__builtin_amdgcn_global_load_async_to_lds_b128)
#define HAVE_ASYNC_LDS 1
#endif

__device__ __forceinline__ void wait_async_lds() {
#if defined(HAVE_ASYNC_LDS)
#if __has_builtin(__builtin_amdgcn_s_wait_asynccnt)
  __builtin_amdgcn_s_wait_asynccnt(0);
#else
  asm volatile("s_wait_asynccnt 0" ::: "memory");
#endif
#endif
}

__device__ __forceinline__ void copy16_g2lds(const bf16* g, bf16* l) {
#if defined(HAVE_ASYNC_LDS)
  __builtin_amdgcn_global_load_async_to_lds_b128(
      (__attribute__((address_space(1))) b128v*)g,
      (__attribute__((address_space(3))) b128v*)l, 0, 0);
#else
  *(float4*)l = *(const float4*)g;
#endif
}

// ---------------------------------------------------------------------------
// WMMA helpers: D(16x16,f32) += A(16x32,bf16) x B(32x16,bf16)
// A fragment (lane): row = lane&15, elems 0..7 -> K = kb + (lane>>4)*8 + e,
//                    elems 8..15 -> K = kb + 16 + (lane>>4)*8 + (e-8)
// B fragment (lane): col = lane&15, same K pattern -> K-major source gives
//                    two contiguous 16B loads per lane.
// C/D (lane):        elem r -> M = r + 8*(lane>>4), N = lane&15
// ---------------------------------------------------------------------------
__device__ __forceinline__ v8f wmma_bf16(v16bf a, v16bf b, v8f c) {
  return __builtin_amdgcn_wmma_f32_16x16x32_bf16(false, a, false, b,
                                                 (short)0, c, false, false);
}

__device__ __forceinline__ v16bf load_frag(const bf16* __restrict__ base,
                                           int stride, int row, int kb, int lane) {
  const bf16* p = base + (size_t)row * stride + kb + ((lane >> 4) << 3);
  v16bf f;
  ((float4*)&f)[0] = ((const float4*)p)[0];
  ((float4*)&f)[1] = ((const float4*)(p + 16))[0];
  return f;
}

__device__ __forceinline__ v16bf lds_frag(const bf16* p, int lane) {
  int o = (lane >> 4) << 3;
  v16bf f;
  ((float4*)&f)[0] = *(const float4*)(p + o);
  ((float4*)&f)[1] = *(const float4*)(p + 16 + o);
  return f;
}

__device__ __forceinline__ float gelu_exact(float v) {
  return 0.5f * v * (1.0f + erff(v * 0.70710678118654752f));
}

// ---------------------------------------------------------------------------
// fp32 -> bf16 cast
// ---------------------------------------------------------------------------
__global__ void k_cast(const float* __restrict__ in, bf16* __restrict__ out, int n) {
  int i = blockIdx.x * blockDim.x + threadIdx.x;
  if (i < n) out[i] = (bf16)in[i];
}

// ---------------------------------------------------------------------------
// LayerNorm over D=512, one 256-thread block per row, bf16 out
// ---------------------------------------------------------------------------
__global__ __launch_bounds__(256) void k_layernorm(
    const float* __restrict__ x, const float* __restrict__ g,
    const float* __restrict__ be, bf16* __restrict__ out) {
  int row = blockIdx.x, tid = threadIdx.x;
  const float* xr = x + (size_t)row * D_MODEL;
  float v0 = xr[tid], v1 = xr[tid + 256];
  __shared__ float red[256];
  red[tid] = v0 + v1;
  __syncthreads();
  for (int st = 128; st > 0; st >>= 1) {
    if (tid < st) red[tid] += red[tid + st];
    __syncthreads();
  }
  float mu = red[0] * (1.0f / D_MODEL);
  __syncthreads();
  float d0 = v0 - mu, d1 = v1 - mu;
  red[tid] = d0 * d0 + d1 * d1;
  __syncthreads();
  for (int st = 128; st > 0; st >>= 1) {
    if (tid < st) red[tid] += red[tid + st];
    __syncthreads();
  }
  float rs = rsqrtf(red[0] * (1.0f / D_MODEL) + 1e-5f);
  bf16* o = out + (size_t)row * D_MODEL;
  o[tid]       = (bf16)(d0 * rs * g[tid]       + be[tid]);
  o[tid + 256] = (bf16)(d1 * rs * g[tid + 256] + be[tid + 256]);
}

// ---------------------------------------------------------------------------
// GEMM: C[M,N] = A[M,K](bf16, K-major) x W[N,K](bf16, K-major)
// 8 waves/block, wave = 16x64 tile. W tile (64x32) staged in LDS via async
// global->LDS copies, double-buffered, padded to stride 40 (conflict-free
// ds_load_b128). Fused templated epilogue.
// ---------------------------------------------------------------------------
#define WSTRIDE 40
template <bool GELU, bool BIAS, bool RES, bool OUTB>
__global__ __launch_bounds__(256) void k_gemm(
    const bf16* __restrict__ A, const bf16* __restrict__ W, int N, int K,
    const float* __restrict__ bias, const float* __restrict__ resid,
    bf16* __restrict__ outB, float* __restrict__ outF) {
  int tid = threadIdx.x;
  int lane = tid & 31;
  int wave = tid >> 5;
  int rr = lane & 15, hi = lane >> 4;
  int m0 = blockIdx.x * 128 + wave * 16;
  int n0 = blockIdx.y * 64;

  __shared__ bf16 Wt[2][64 * WSTRIDE];
  int nl = tid >> 2;            // 0..63
  int kl = (tid & 3) << 3;      // 0,8,16,24

  // prefill buffer 0
  copy16_g2lds(W + (size_t)(n0 + nl) * K + kl, &Wt[0][nl * WSTRIDE + kl]);
  wait_async_lds();

  v8f acc[4] = {};
  for (int kb = 0; kb < K; kb += 32) {
    int buf = (kb >> 5) & 1;
    __syncthreads();
    if (kb + 32 < K)
      copy16_g2lds(W + (size_t)(n0 + nl) * K + kb + 32 + kl,
                   &Wt[buf ^ 1][nl * WSTRIDE + kl]);
    if (kb + 64 < K)
      __builtin_prefetch(A + (size_t)(m0 + rr) * K + kb + 64, 0, 0);
    v16bf a = load_frag(A, K, m0 + rr, kb, lane);
#pragma unroll
    for (int t = 0; t < 4; ++t) {
      v16bf b = lds_frag(&Wt[buf][(t * 16 + rr) * WSTRIDE], lane);
      acc[t] = wmma_bf16(a, b, acc[t]);
    }
    wait_async_lds();
  }

#pragma unroll
  for (int t = 0; t < 4; ++t) {
    int n = n0 + t * 16 + rr;
    float bv = BIAS ? bias[n] : 0.0f;
#pragma unroll
    for (int r = 0; r < 8; ++r) {
      int m = m0 + r + 8 * hi;
      float v = acc[t][r] + bv;
      if (GELU) v = gelu_exact(v);
      size_t idx = (size_t)m * N + n;
      if (RES) v += resid[idx];
      if (OUTB) outB[idx] = (bf16)v;
      else      outF[idx] = v;
    }
  }
}

// ---------------------------------------------------------------------------
// V transpose: [B][L][D] -> [B][D][L], bf16, 32x32 LDS tiles
// ---------------------------------------------------------------------------
__global__ void k_transpose(const bf16* __restrict__ in, bf16* __restrict__ out,
                            int L, int D) {
  __shared__ bf16 tile[32][33];
  int b = blockIdx.z, l0 = blockIdx.x * 32, d0 = blockIdx.y * 32;
  for (int i = threadIdx.y; i < 32; i += 8)
    tile[i][threadIdx.x] = in[((size_t)b * L + l0 + i) * D + d0 + threadIdx.x];
  __syncthreads();
  for (int i = threadIdx.y; i < 32; i += 8)
    out[((size_t)b * D + d0 + i) * L + l0 + threadIdx.x] = tile[threadIdx.x][i];
}

// ---------------------------------------------------------------------------
// Flash attention. One wave owns 16 queries x full D=512.
//   CAUSAL=1: masked, writes Ob (bf16) = softmax(QK^T/sqrt(D)) V
//   CAUSAL=0: Oadd += gate * softmax(QK^T * scale/T) V   (re-attention)
// Q fragments register-resident; P staged through a per-wave LDS tile to
// convert C-layout fp32 -> A-layout bf16 (DS ops are in-order per wave).
// ---------------------------------------------------------------------------
template <bool CAUSAL>
__global__ __launch_bounds__(128, 1) void k_flash(
    const bf16* __restrict__ Q, const bf16* __restrict__ Km,
    const bf16* __restrict__ Vt, bf16* __restrict__ Ob,
    float* __restrict__ Oadd, const float* __restrict__ scal, int L) {
  int lane = threadIdx.x & 31;
  int wave = threadIdx.x >> 5;
  int rr = lane & 15, hi = lane >> 4;
  int b = blockIdx.y;
  int q0 = (blockIdx.x * 4 + wave) * 16;

  const bf16* Qb = Q  + (size_t)b * L * D_MODEL;
  const bf16* Kb = Km + (size_t)b * L * D_MODEL;
  const bf16* Vb = Vt + (size_t)b * D_MODEL * L;

  float scale = INV_SQRT_D;
  float gate = 1.0f;
  if (!CAUSAL) { scale /= scal[0]; gate = scal[1]; }

  v16bf qf[16];
#pragma unroll
  for (int c = 0; c < 16; ++c)
    qf[c] = load_frag(Qb, D_MODEL, q0 + rr, c * 32, lane);

  v8f o[NT];
#pragma unroll
  for (int t = 0; t < NT; ++t)
    o[t] = (v8f){0.f, 0.f, 0.f, 0.f, 0.f, 0.f, 0.f, 0.f};

  float rmax[8], rsum[8];
#pragma unroll
  for (int r = 0; r < 8; ++r) { rmax[r] = -1e30f; rsum[r] = 0.0f; }

  __shared__ float sS[4][16][32];
  float (*Sw)[32] = sS[wave];

  int kend = CAUSAL ? (q0 + 16) : L;
  for (int j0 = 0; j0 < kend; j0 += 32) {
    if (j0 + 32 < kend)
      __builtin_prefetch(Kb + (size_t)(j0 + 32 + rr) * D_MODEL, 0, 0);
    v8f s0 = (v8f){0.f, 0.f, 0.f, 0.f, 0.f, 0.f, 0.f, 0.f};
    v8f s1 = (v8f){0.f, 0.f, 0.f, 0.f, 0.f, 0.f, 0.f, 0.f};
#pragma unroll
    for (int c = 0; c < 16; ++c) {
      v16bf kf0 = load_frag(Kb, D_MODEL, j0 + rr,      c * 32, lane);
      v16bf kf1 = load_frag(Kb, D_MODEL, j0 + 16 + rr, c * 32, lane);
      s0 = wmma_bf16(qf[c], kf0, s0);
      s1 = wmma_bf16(qf[c], kf1, s1);
    }
    float e0a[8], e1a[8];
#pragma unroll
    for (int r = 0; r < 8; ++r) {
      int m = q0 + r + 8 * hi;
      float a0 = s0[r] * scale;
      float a1 = s1[r] * scale;
      if (CAUSAL) {
        if (j0 + rr > m)      a0 = -1e30f;
        if (j0 + 16 + rr > m) a1 = -1e30f;
      }
      float mx = fmaxf(a0, a1);
#pragma unroll
      for (int off = 1; off < 16; off <<= 1)
        mx = fmaxf(mx, __shfl_xor(mx, off, 32));
      float nm = fmaxf(rmax[r], mx);
      float corr = __expf(rmax[r] - nm);
      rmax[r] = nm;
      float e0 = __expf(a0 - nm);
      float e1 = __expf(a1 - nm);
      float ps = e0 + e1;
#pragma unroll
      for (int off = 1; off < 16; off <<= 1)
        ps += __shfl_xor(ps, off, 32);
      rsum[r] = rsum[r] * corr + ps;
#pragma unroll
      for (int t = 0; t < NT; ++t) o[t][r] *= corr;
      e0a[r] = e0;
      e1a[r] = e1;
    }
    // C-layout -> LDS -> A-layout bf16 (per-wave tile, DS in-order per wave)
#pragma unroll
    for (int r = 0; r < 8; ++r) {
      Sw[r + 8 * hi][rr]      = e0a[r];
      Sw[r + 8 * hi][16 + rr] = e1a[r];
    }
    asm volatile("s_wait_dscnt 0" ::: "memory");
    v16bf pf;
#pragma unroll
    for (int e = 0; e < 8; ++e) {
      pf[e]     = (bf16)Sw[rr][hi * 8 + e];
      pf[e + 8] = (bf16)Sw[rr][16 + hi * 8 + e];
    }
#pragma unroll
    for (int t = 0; t < NT; ++t) {
      v16bf vf = load_frag(Vb, L, t * 16 + rr, j0, lane);
      o[t] = wmma_bf16(pf, vf, o[t]);
    }
  }

#pragma unroll
  for (int t = 0; t < NT; ++t) {
    int d = t * 16 + rr;
#pragma unroll
    for (int r = 0; r < 8; ++r) {
      int m = q0 + r + 8 * hi;
      float val = o[t][r] / rsum[r];
      size_t idx = ((size_t)b * L + m) * D_MODEL + d;
      if (CAUSAL) Ob[idx] = (bf16)val;
      else        Oadd[idx] += gate * val;
    }
  }
}

// ---------------------------------------------------------------------------
// Entropy of softmax of the last query row, per batch (streaming LSE).
// H = log(S0) - S1/S0,  S0 = sum e^{a-m},  S1 = sum (a-m) e^{a-m}
// ---------------------------------------------------------------------------
__global__ __launch_bounds__(256) void k_lastrow(
    const bf16* __restrict__ Q, const bf16* __restrict__ Km,
    float* __restrict__ H, int L) {
  int b = blockIdx.x, tid = threadIdx.x;
  const bf16* q  = Q  + ((size_t)b * L + (L - 1)) * D_MODEL;
  const bf16* Kb = Km + (size_t)b * L * D_MODEL;
  float mmax = -1e30f, s0 = 0.0f, s1 = 0.0f;
  for (int m = tid; m < L; m += 256) {
    float dot = 0.0f;
    for (int d = 0; d < D_MODEL; ++d)
      dot += (float)q[d] * (float)Kb[(size_t)m * D_MODEL + d];
    float a = dot * INV_SQRT_D;
    if (a > mmax) {
      float dlt = a - mmax;
      float c = __expf(-dlt);
      s1 = (s1 - dlt * s0) * c;
      s0 = s0 * c + 1.0f;
      mmax = a;
    } else {
      float e = __expf(a - mmax);
      s0 += e;
      s1 += (a - mmax) * e;
    }
  }
  __shared__ float sm[256], sa[256], sb[256];
  sm[tid] = mmax; sa[tid] = s0; sb[tid] = s1;
  __syncthreads();
  for (int st = 128; st > 0; st >>= 1) {
    if (tid < st) {
      float m1 = sm[tid], m2 = sm[tid + st];
      float nm = fmaxf(m1, m2);
      float d1 = nm - m1, d2 = nm - m2;
      float c1 = __expf(-d1), c2 = __expf(-d2);
      sb[tid] = (sb[tid] - d1 * sa[tid]) * c1 + (sb[tid + st] - d2 * sa[tid + st]) * c2;
      sa[tid] = sa[tid] * c1 + sa[tid + st] * c2;
      sm[tid] = nm;
    }
    __syncthreads();
  }
  if (tid == 0) H[b] = __logf(sa[0]) - sb[0] / sa[0];
}

// ---------------------------------------------------------------------------
// K_rep[b][d] = mean_l K[b][l][d]
// ---------------------------------------------------------------------------
__global__ void k_kmean(const bf16* __restrict__ Km, float* __restrict__ Krep, int L) {
  int b = blockIdx.y;
  int d = blockIdx.x * blockDim.x + threadIdx.x;
  float s = 0.0f;
  for (int l = 0; l < L; ++l)
    s += (float)Km[((size_t)b * L + l) * D_MODEL + d];
  Krep[b * D_MODEL + d] = s * (1.0f / L);
}

// ---------------------------------------------------------------------------
// Scalars: T = 0.6 + 1.0*Hm ; sim_th = 0.9 - 0.4*Hm ; hit gate. One wave.
// ---------------------------------------------------------------------------
__global__ __launch_bounds__(32) void k_scalars(
    const float* __restrict__ H, const bf16* __restrict__ Q,
    const float* __restrict__ Krep, float* __restrict__ scal, int L) {
  int lane = threadIdx.x;
  float h0 = H[0], h1 = H[1];
  float hmin = fminf(h0, h1), hmax = fmaxf(h0, h1);
  float inv = 1.0f / (hmax - hmin + 1e-9f);
  float Hm = 0.5f * ((h0 - hmin) * inv + (h1 - hmin) * inv);
  float T = 0.6f + 1.0f * Hm;
  float sim_th = 0.9f - 0.4f * Hm;
  float qq = 0, d0 = 0, n0 = 0, d1 = 0, n1 = 0;
  for (int d = lane; d < D_MODEL; d += 32) {
    float qv = 0.5f * ((float)Q[((size_t)0 * L + L - 1) * D_MODEL + d] +
                       (float)Q[((size_t)1 * L + L - 1) * D_MODEL + d]);
    float k0 = Krep[d], k1 = Krep[D_MODEL + d];
    qq += qv * qv; d0 += qv * k0; n0 += k0 * k0; d1 += qv * k1; n1 += k1 * k1;
  }
#pragma unroll
  for (int off = 16; off > 0; off >>= 1) {
    qq += __shfl_xor(qq, off, 32);
    d0 += __shfl_xor(d0, off, 32);
    n0 += __shfl_xor(n0, off, 32);
    d1 += __shfl_xor(d1, off, 32);
    n1 += __shfl_xor(n1, off, 32);
  }
  if (lane == 0) {
    float qn = fmaxf(sqrtf(qq), 1e-8f);
    float c0 = d0 / (qn * fmaxf(sqrtf(n0), 1e-8f));
    float c1 = d1 / (qn * fmaxf(sqrtf(n1), 1e-8f));
    float s = 0.5f * (c0 + c1);
    scal[0] = T;
    scal[1] = (s >= sim_th) ? 1.0f : 0.0f;
  }
}

// ---------------------------------------------------------------------------
extern "C" void kernel_launch(void* const* d_in, const int* in_sizes, int n_in,
                              void* d_out, int out_size, void* d_ws, size_t ws_size,
                              hipStream_t stream) {
  (void)in_sizes; (void)n_in; (void)out_size; (void)ws_size;
  const int B = 2, L = 4096, D = 512, F = 2048, M = B * L;

  const float* x     = (const float*)d_in[0];
  const float* Wq    = (const float*)d_in[1];
  const float* Wk    = (const float*)d_in[2];
  const float* Wv    = (const float*)d_in[3];
  const float* Wp    = (const float*)d_in[4];
  const float* W1    = (const float*)d_in[5];
  const float* b1    = (const float*)d_in[6];
  const float* W2    = (const float*)d_in[7];
  const float* b2    = (const float*)d_in[8];
  const float* gamma = (const float*)d_in[9];
  const float* beta  = (const float*)d_in[10];
  float* out = (float*)d_out;

  size_t off = 0;
  char* base = (char*)d_ws;
  auto take = [&](size_t bytes) -> char* {
    char* p = base + off;
    off = (off + bytes + 255) & ~(size_t)255;
    return p;
  };
  bf16*  xn   = (bf16*)take((size_t)M * D * 2);
  bf16*  Qb   = (bf16*)take((size_t)M * D * 2);
  bf16*  Kb   = (bf16*)take((size_t)M * D * 2);
  bf16*  Vb   = (bf16*)take((size_t)M * D * 2);
  bf16*  Vt   = (bf16*)take((size_t)M * D * 2);
  bf16*  aO   = (bf16*)take((size_t)M * D * 2);
  float* ctx  = (float*)take((size_t)M * D * 4);
  bf16*  hn   = (bf16*)take((size_t)M * D * 2);
  bf16*  hbuf = (bf16*)take((size_t)M * F * 2);
  bf16*  WqB  = (bf16*)take((size_t)D * D * 2);
  bf16*  WkB  = (bf16*)take((size_t)D * D * 2);
  bf16*  WvB  = (bf16*)take((size_t)D * D * 2);
  bf16*  WpB  = (bf16*)take((size_t)D * D * 2);
  bf16*  W1B  = (bf16*)take((size_t)F * D * 2);
  bf16*  W2B  = (bf16*)take((size_t)D * F * 2);
  float* Hb   = (float*)take(2 * sizeof(float));
  float* Krep = (float*)take((size_t)B * D * sizeof(float));
  float* scal = (float*)take(2 * sizeof(float));

  // Weights -> bf16
  k_cast<<<dim3((D * D + 255) / 256), 256, 0, stream>>>(Wq, WqB, D * D);
  k_cast<<<dim3((D * D + 255) / 256), 256, 0, stream>>>(Wk, WkB, D * D);
  k_cast<<<dim3((D * D + 255) / 256), 256, 0, stream>>>(Wv, WvB, D * D);
  k_cast<<<dim3((D * D + 255) / 256), 256, 0, stream>>>(Wp, WpB, D * D);
  k_cast<<<dim3((F * D + 255) / 256), 256, 0, stream>>>(W1, W1B, F * D);
  k_cast<<<dim3((D * F + 255) / 256), 256, 0, stream>>>(W2, W2B, D * F);

  // LN1
  k_layernorm<<<M, 256, 0, stream>>>(x, gamma, beta, xn);

  // Q, K, V  (bf16 out)
  dim3 gq(M / 128, D / 64);
  k_gemm<false,false,false,true><<<gq, 256, 0, stream>>>(xn, WqB, D, D, nullptr, nullptr, Qb, nullptr);
  k_gemm<false,false,false,true><<<gq, 256, 0, stream>>>(xn, WkB, D, D, nullptr, nullptr, Kb, nullptr);
  k_gemm<false,false,false,true><<<gq, 256, 0, stream>>>(xn, WvB, D, D, nullptr, nullptr, Vb, nullptr);

  // V^T for the P*V matmul
  k_transpose<<<dim3(L / 32, D / 32, B), dim3(32, 8), 0, stream>>>(Vb, Vt, L, D);

  // Causal flash attention -> aO (bf16)
  k_flash<true><<<dim3(L / 64, B), 128, 0, stream>>>(Qb, Kb, Vt, aO, nullptr, nullptr, L);

  // ctx = x + aO @ Wp^T  (fp32)
  k_gemm<false,false,true,false><<<gq, 256, 0, stream>>>(aO, WpB, D, D, nullptr, x, nullptr, ctx);

  // LN2 -> hn (bf16)
  k_layernorm<<<M, 256, 0, stream>>>(ctx, gamma, beta, hn);

  // h = gelu(hn @ W1^T + b1)  (bf16)
  k_gemm<true,true,false,true><<<dim3(M / 128, F / 64), 256, 0, stream>>>(hn, W1B, F, D, b1, nullptr, hbuf, nullptr);

  // out = ctx + h @ W2^T + b2  (fp32)
  k_gemm<false,true,true,false><<<dim3(M / 128, D / 64), 256, 0, stream>>>(hbuf, W2B, D, F, b2, ctx, nullptr, out);

  // Entropy / cosine-sim scalars (device-side, graph-safe)
  k_lastrow<<<B, 256, 0, stream>>>(Qb, Kb, Hb, L);
  k_kmean<<<dim3(D / 256, B), 256, 0, stream>>>(Kb, Krep, L);
  k_scalars<<<1, 32, 0, stream>>>(Hb, Qb, Krep, scal, L);

  // Re-attention (folded: softmax(QK^T/(sqrt(D)*T)) @ V), gated add into out
  k_flash<false><<<dim3(L / 64, B), 128, 0, stream>>>(Qb, Kb, Vt, nullptr, out, scal, L);
}
